// GenLSTM_36429912605350
// MI455X (gfx1250) — matrix operand order repaired and checked by reading, not compiled, not run on hardware
//
#include <hip/hip_runtime.h>
#include <hip/hip_bf16.h>

// ---------------------------------------------------------------------------
// Autoregressive LSTM generator for MI455X (gfx1250, wave32, WMMA).
// bf16 WMMA (16x16x32) with f32 accumulate.  Gates GEMM is LDS-staged and
// double-buffered via CDNA5 global_load_async_to_lds_b128 (ASYNCcnt), so L2
// latency overlaps the WMMA pipe, weights are read from L2 once per block,
// and there is no VGPR bounce for the staging copies.
// ---------------------------------------------------------------------------

typedef __bf16 bf16;
typedef __attribute__((ext_vector_type(8)))  __bf16 bf16x8;
typedef __attribute__((ext_vector_type(16))) __bf16 bf16x16;
typedef __attribute__((ext_vector_type(8)))  float  f32x8;
typedef __attribute__((ext_vector_type(4)))  int    v4i;

#define HID    512
#define NBATCH 1024
#define NSTEP  512
#define SEQD   64
#define INSZ   129      // SEQ_DIM + NOISE_DIM + 1
#define XPAD   160      // INSZ padded to multiple of 32
#define KTOT   672      // XPAD + HID  (concatenated K dimension)
#define GATES  2048     // 4 * HID
#define NCHUNK (KTOT / 32)   // 21
#define APAD   40       // LDS row pitch in bf16 (80B) -> conflict-free frags

union FragB16 { bf16x16 v; bf16x8 h[2]; };

__device__ __forceinline__ float sigmf(float x)   { return 1.0f / (1.0f + __expf(-x)); }
__device__ __forceinline__ float tanhfast(float x){ return 1.0f - 2.0f / (__expf(2.0f * x) + 1.0f); }

#if defined(__has_builtin)
#if __has_builtin(__builtin_amdgcn_global_load_async_to_lds_b128) && \
    __has_builtin(__builtin_amdgcn_s_wait_asynccnt)
#define HAVE_ASYNC_LDS 1
#endif
#endif

#if defined(HAVE_ASYNC_LDS)
typedef __attribute__((address_space(1))) v4i* gv4i_p;
typedef __attribute__((address_space(3))) v4i* lv4i_p;
#endif

// ---------------------------------------------------------------------------
// Gates GEMM + LSTM cell.  Block = 64 batch rows x 64 hidden cols (=256 gate
// cols), 512 threads = 16 waves.  Wave (wr = wv&3, wc = wv>>2) computes
// 16 rows x (4 gates x 16 hidden cols) = 4 WMMA tiles, so i/f/g/o of a hidden
// unit live in the same lane -> register-resident cell update.
// A/B chunks land in double-buffered LDS via async copies; chunk c+1's copies
// are issued before chunk c's WMMAs, and each chunk costs one
// s_wait_asynccnt + one barrier.  All fragment ds_loads of a chunk are issued
// together so the scheduler can use partial dscnt waits.
// ---------------------------------------------------------------------------
__global__ __launch_bounds__(512)
void lstm_gates(const bf16* __restrict__ Xcur,   // [NBATCH][KTOT] bf16 (x|h)
                bf16*       __restrict__ Xnxt,   // next-step buffer (h section)
                const bf16* __restrict__ Wcat,   // [GATES][KTOT] bf16
                const float* __restrict__ bias,  // [GATES] f32 (b_ih + b_hh)
                float*      __restrict__ Cst,    // [NBATCH][HID] f32 cell state
                float*      __restrict__ Hout,   // final-h f32 region of d_out
                int write_state)
{
    __shared__ __attribute__((aligned(16))) bf16 Abuf[2][64  * APAD];   // 10 KB
    __shared__ __attribute__((aligned(16))) bf16 Bbuf[2][256 * APAD];   // 40 KB

    const int tid  = threadIdx.x;
    const int lane = tid & 31;
    const int wv   = tid >> 5;
    const int wr   = wv & 3;      // 16-row strip
    const int wc   = wv >> 2;     // 16-hidden-col quarter
    const int lo   = lane & 15;
    const int hi   = lane >> 4;

    const int m0  = blockIdx.x * 64;             // batch rows of this block
    const int hb  = blockIdx.y * 64;             // hidden cols of this block

    // ---- staging assignments: each thread owns a 32B slice per chunk ----
    // B: 256 rows x 32 k-elems (16KB); thread = (row, 16-elem half)
    const int brow  = tid >> 1;
    const int bhalf = tid & 1;
    const int bng   = (brow >> 6) * HID + hb + (brow & 63);     // global gate row
    const bf16* srcB = Wcat + (size_t)bng * KTOT + bhalf * 16;
    const int   dstB = brow * APAD + bhalf * 16;
    // A: 64 rows x 32 k-elems (4KB); first 128 threads, thread = (row, half)
    const int arow  = tid >> 1;          // valid when tid < 128
    const int ahalf = tid & 1;
    const bf16* srcA = Xcur + (size_t)(m0 + arow) * KTOT + ahalf * 16;
    const int   dstA = arow * APAD + ahalf * 16;

    // ---- per-wave fragment offsets in LDS ----
    const int aoff = (wr * 16 + lo) * APAD + hi * 8;   // + {0,16} for the 2 halves
    int boff[4];
#pragma unroll
    for (int g = 0; g < 4; ++g)
        boff[g] = (g * 64 + wc * 16 + lo) * APAD + hi * 8;

    f32x8 acc[4] = {};

#if defined(HAVE_ASYNC_LDS)
    // -------- async staging: global -> LDS directly, tracked by ASYNCcnt ----
    auto issue_async = [&](int buf, int k0) {
        gv4i_p gb = (gv4i_p)(srcB + k0);
        lv4i_p lb = (lv4i_p)&Bbuf[buf][dstB];
        __builtin_amdgcn_global_load_async_to_lds_b128(gb, lb, 0, 0);
        __builtin_amdgcn_global_load_async_to_lds_b128(gb, lb, 16, 0);
        if (tid < 128) {
            gv4i_p ga = (gv4i_p)(srcA + k0);
            lv4i_p la = (lv4i_p)&Abuf[buf][dstA];
            __builtin_amdgcn_global_load_async_to_lds_b128(ga, la, 0, 0);
            __builtin_amdgcn_global_load_async_to_lds_b128(ga, la, 16, 0);
        }
    };

    issue_async(0, 0);
    for (int c = 0; c < NCHUNK; ++c) {
        const int buf = c & 1;
        __builtin_amdgcn_s_wait_asynccnt(0);   // this wave's copies for chunk c done
        __syncthreads();                       // all waves' copies visible; prior
                                               // readers of buf^1 are finished
        if (c + 1 < NCHUNK) issue_async(buf ^ 1, (c + 1) * 32);

        // Issue all fragment loads together -> partial dscnt waits.
        FragB16 a, bfr[4];
        a.h[0] = *(const bf16x8*)&Abuf[buf][aoff];
        a.h[1] = *(const bf16x8*)&Abuf[buf][aoff + 16];
#pragma unroll
        for (int g = 0; g < 4; ++g) {
            bfr[g].h[0] = *(const bf16x8*)&Bbuf[buf][boff[g]];
            bfr[g].h[1] = *(const bf16x8*)&Bbuf[buf][boff[g] + 16];
        }
#pragma unroll
        for (int g = 0; g < 4; ++g)
            acc[g] = __builtin_amdgcn_wmma_f32_16x16x32_bf16(
                         false, a.v, false, bfr[g].v, (short)0, acc[g], false, false);
    }
#else
    // -------- fallback: global -> VGPR -> LDS staging ----------------------
    bf16x8 sB0, sB1, sA0, sA1;
    auto stage_load = [&](int k0) {
        sB0 = *(const bf16x8*)(srcB + k0);
        sB1 = *(const bf16x8*)(srcB + k0 + 8);
        if (tid < 128) {
            sA0 = *(const bf16x8*)(srcA + k0);
            sA1 = *(const bf16x8*)(srcA + k0 + 8);
        }
    };
    auto stage_store = [&](int buf) {
        *(bf16x8*)&Bbuf[buf][dstB]     = sB0;
        *(bf16x8*)&Bbuf[buf][dstB + 8] = sB1;
        if (tid < 128) {
            *(bf16x8*)&Abuf[buf][dstA]     = sA0;
            *(bf16x8*)&Abuf[buf][dstA + 8] = sA1;
        }
    };
    stage_load(0);
    stage_store(0);
    __syncthreads();
    for (int c = 0; c < NCHUNK; ++c) {
        const int buf = c & 1;
        if (c + 1 < NCHUNK) stage_load((c + 1) * 32);
        FragB16 a, bfr[4];
        a.h[0] = *(const bf16x8*)&Abuf[buf][aoff];
        a.h[1] = *(const bf16x8*)&Abuf[buf][aoff + 16];
#pragma unroll
        for (int g = 0; g < 4; ++g) {
            bfr[g].h[0] = *(const bf16x8*)&Bbuf[buf][boff[g]];
            bfr[g].h[1] = *(const bf16x8*)&Bbuf[buf][boff[g] + 16];
        }
#pragma unroll
        for (int g = 0; g < 4; ++g)
            acc[g] = __builtin_amdgcn_wmma_f32_16x16x32_bf16(
                         false, a.v, false, bfr[g].v, (short)0, acc[g], false, false);
        __syncthreads();
        if (c + 1 < NCHUNK) {
            stage_store(buf ^ 1);
            __syncthreads();
        }
    }
#endif

    // ---- epilogue: elementwise LSTM cell, fully in registers ----
    const int j = hb + wc * 16 + lo;                 // hidden index
    const float bi = bias[0 * HID + j];
    const float bf_ = bias[1 * HID + j];
    const float bg = bias[2 * HID + j];
    const float bo = bias[3 * HID + j];
#pragma unroll
    for (int r = 0; r < 8; ++r) {
        const int m = m0 + wr * 16 + hi * 8 + r;
        float iv = acc[0][r] + bi;
        float fv = acc[1][r] + bf_;
        float gv = acc[2][r] + bg;
        float ov = acc[3][r] + bo;
        const size_t ci = (size_t)m * HID + j;
        float c_prev = Cst[ci];
        float c_new  = sigmf(fv) * c_prev + sigmf(iv) * tanhfast(gv);
        float h_new  = sigmf(ov) * tanhfast(c_new);
        Cst[ci] = c_new;
        Xnxt[(size_t)m * KTOT + XPAD + j] = (bf16)h_new;
        if (write_state) Hout[ci] = h_new;
    }
}

// ---------------------------------------------------------------------------
// Output projection: y = h_new @ W_out^T + b_out.  W_out is 64KB and
// L2-resident; fragments come straight from global with explicit register
// double-buffering so loads overlap WMMAs.
// ---------------------------------------------------------------------------
__global__ __launch_bounds__(256)
void lstm_out(bf16*       __restrict__ Xnxt,     // read h (cols XPAD..), write last
              const bf16* __restrict__ WoutB,    // [SEQD][HID] bf16
              const float* __restrict__ b_out,   // [SEQD]
              float*      __restrict__ gen,      // [NBATCH][NSTEP][SEQD]
              int step)
{
    const int lane = threadIdx.x & 31;
    const int wv   = threadIdx.x >> 5;
    const int wr   = wv & 3;
    const int wc   = wv >> 2;
    const int lo   = lane & 15;
    const int hi   = lane >> 4;

    const int m0 = blockIdx.x * 64 + wr * 16;

    const bf16* pA = Xnxt + (size_t)(m0 + lo) * KTOT + XPAD + hi * 8;
    int boff[2];
#pragma unroll
    for (int h = 0; h < 2; ++h)
        boff[h] = (wc * 32 + h * 16 + lo) * HID + hi * 8;

    f32x8 acc[2] = {};
    FragB16 aC, bC[2], aN, bN[2];

    aC.h[0] = *(const bf16x8*)(pA);
    aC.h[1] = *(const bf16x8*)(pA + 16);
#pragma unroll
    for (int h = 0; h < 2; ++h) {
        bC[h].h[0] = *(const bf16x8*)(WoutB + boff[h]);
        bC[h].h[1] = *(const bf16x8*)(WoutB + boff[h] + 16);
    }

    for (int k0 = 32; k0 < HID; k0 += 32) {
        aN.h[0] = *(const bf16x8*)(pA + k0);
        aN.h[1] = *(const bf16x8*)(pA + k0 + 16);
#pragma unroll
        for (int h = 0; h < 2; ++h) {
            bN[h].h[0] = *(const bf16x8*)(WoutB + boff[h] + k0);
            bN[h].h[1] = *(const bf16x8*)(WoutB + boff[h] + k0 + 16);
        }
#pragma unroll
        for (int h = 0; h < 2; ++h)
            acc[h] = __builtin_amdgcn_wmma_f32_16x16x32_bf16(
                         false, aC.v, false, bC[h].v, (short)0, acc[h], false, false);
        aC = aN;
        bC[0] = bN[0];
        bC[1] = bN[1];
    }
#pragma unroll
    for (int h = 0; h < 2; ++h)
        acc[h] = __builtin_amdgcn_wmma_f32_16x16x32_bf16(
                     false, aC.v, false, bC[h].v, (short)0, acc[h], false, false);

#pragma unroll
    for (int h = 0; h < 2; ++h) {
        const int n = wc * 32 + h * 16 + lo;
        const float bo = b_out[n];
#pragma unroll
        for (int r = 0; r < 8; ++r) {
            const int m = m0 + hi * 8 + r;
            float y = acc[h][r] + bo;
            gen[(size_t)m * (NSTEP * SEQD) + (size_t)step * SEQD + n] = y;
            Xnxt[(size_t)m * KTOT + n] = (bf16)y;
        }
    }
}

// ---------------------------------------------------------------------------
// One-time setup kernels
// ---------------------------------------------------------------------------
__global__ void init_wcat(const float* __restrict__ W_ih, const float* __restrict__ W_hh,
                          const float* __restrict__ b_ih, const float* __restrict__ b_hh,
                          bf16* __restrict__ Wcat, float* __restrict__ bias)
{
    int idx = blockIdx.x * 256 + threadIdx.x;
    if (idx >= GATES * KTOT) return;
    int n = idx / KTOT;
    int k = idx - n * KTOT;
    float v = 0.0f;
    if (k < INSZ)       v = W_ih[n * INSZ + k];
    else if (k >= XPAD) v = W_hh[n * HID + (k - XPAD)];
    Wcat[idx] = (bf16)v;
    if (k == 0) bias[n] = b_ih[n] + b_hh[n];
}

__global__ void init_wout(const float* __restrict__ W_out, bf16* __restrict__ WoutB)
{
    int idx = blockIdx.x * 256 + threadIdx.x;
    if (idx < SEQD * HID) WoutB[idx] = (bf16)W_out[idx];
}

__global__ void zero_bf16(bf16* __restrict__ p, int n)
{
    int idx = blockIdx.x * 256 + threadIdx.x;
    if (idx < n) p[idx] = (bf16)0.0f;
}

__global__ void zero_f32(float* __restrict__ p, int n)
{
    int idx = blockIdx.x * 256 + threadIdx.x;
    if (idx < n) p[idx] = 0.0f;
}

// Per-step: write noise_t and dt_t (bf16) into the current X buffer.
__global__ void prep_x(bf16* __restrict__ Xcur, const float* __restrict__ noise,
                       const float* __restrict__ dts, int step)
{
    int idx = blockIdx.x * 256 + threadIdx.x;
    if (idx >= NBATCH * 65) return;
    int b = idx / 65;
    int c = idx - b * 65;
    float v = (c < 64) ? noise[((size_t)b * NSTEP + step) * 64 + c]
                       : dts[(size_t)b * NSTEP + step];
    Xcur[(size_t)b * KTOT + 64 + c] = (bf16)v;
}

// ---------------------------------------------------------------------------
extern "C" void kernel_launch(void* const* d_in, const int* in_sizes, int n_in,
                              void* d_out, int out_size, void* d_ws, size_t ws_size,
                              hipStream_t stream)
{
    const float* noise = (const float*)d_in[0];
    const float* dts   = (const float*)d_in[1];
    const float* W_ih  = (const float*)d_in[2];
    const float* W_hh  = (const float*)d_in[3];
    const float* b_ih  = (const float*)d_in[4];
    const float* b_hh  = (const float*)d_in[5];
    const float* W_out = (const float*)d_in[6];
    const float* b_out = (const float*)d_in[7];

    float* gen  = (float*)d_out;                         // [1024][512][64]
    float* Hout = gen + (size_t)NBATCH * NSTEP * SEQD;   // [1][1024][512]
    float* Cst  = Hout + (size_t)NBATCH * HID;           // [1][1024][512] (working c)

    // Workspace layout
    bf16* XA    = (bf16*)d_ws;                           // [1024][672]
    bf16* XB    = XA + (size_t)NBATCH * KTOT;            // [1024][672]
    bf16* Wcat  = XB + (size_t)NBATCH * KTOT;            // [2048][672]
    bf16* WoutB = Wcat + (size_t)GATES * KTOT;           // [64][512]
    float* bias = (float*)(WoutB + (size_t)SEQD * HID);  // [2048]

    // One-time setup (runs every launch -> deterministic)
    init_wcat<<<(GATES * KTOT + 255) / 256, 256, 0, stream>>>(W_ih, W_hh, b_ih, b_hh, Wcat, bias);
    init_wout<<<(SEQD * HID + 255) / 256, 256, 0, stream>>>(W_out, WoutB);
    zero_bf16<<<(NBATCH * KTOT + 255) / 256, 256, 0, stream>>>(XA, NBATCH * KTOT);
    zero_bf16<<<(NBATCH * KTOT + 255) / 256, 256, 0, stream>>>(XB, NBATCH * KTOT);
    zero_f32 <<<(NBATCH * HID + 255) / 256, 256, 0, stream>>>(Cst, NBATCH * HID);

    for (int t = 0; t < NSTEP; ++t) {
        bf16* Xcur = (t & 1) ? XB : XA;
        bf16* Xnxt = (t & 1) ? XA : XB;
        prep_x<<<(NBATCH * 65 + 255) / 256, 256, 0, stream>>>(Xcur, noise, dts, t);
        lstm_gates<<<dim3(NBATCH / 64, HID / 64), 512, 0, stream>>>(
            Xcur, Xnxt, Wcat, bias, Cst, Hout, (t == NSTEP - 1) ? 1 : 0);
        lstm_out<<<NBATCH / 64, 256, 0, stream>>>(Xnxt, WoutB, b_out, gen, t);
    }
}
